// Net_83013127897371
// MI455X (gfx1250) — matrix operand order repaired and checked by reading, not compiled, not run on hardware
//
#include <hip/hip_runtime.h>
#include <hip/hip_bf16.h>

typedef __attribute__((ext_vector_type(16))) _Float16 v16h;
typedef __attribute__((ext_vector_type(8)))  _Float16 v8h;
typedef __attribute__((ext_vector_type(8)))  float    v8f;

union F16x16 { v16h v; v8h p[2]; _Float16 h[16]; };

// ---------------------------------------------------------------------------
// Fused conv3x3(SAME) + bias + (leaky)ReLU + maxpool2x2, implicit GEMM on WMMA.
//   in : TIN (f32 or f16) [B,CI,HH,WW]   Wt : f32 [Co,CI,3,3]   bias : f32 [Co]
//   out: f16 [B,Co,HH/2,WW/2]
// Block = 256 threads (8 waves) handles image b and a 128-conv-pixel supertile
// (R=128/WW rows x WW cols). Wave w owns 16 pixels (one WMMA N-tile).
// M = out-channels (NCT tiles of 16), K = CI*9 padded to NK chunks of 32.
// Input halo (R+2)x(WW+2)xCI staged once in LDS (converted to f16 in-flight);
// B fragments gathered directly from the halo in contiguous k order (k = 16*half + i), matching the A
// layout. Weight A-tiles staged once in LDS, f16, padded. The K loop is fully
// unrolled so all tap/channel decodes fold to compile-time constants.
// ---------------------------------------------------------------------------
template <int NCT, int NK, int CI, int HH, int WW, typename TIN>
__global__ __launch_bounds__(256)
void conv3x3_relu_pool(const TIN* __restrict__ in,
                       const float* __restrict__ Wt,
                       const float* __restrict__ bias,
                       _Float16* __restrict__ out,
                       int Co, float slope,
                       const int* __restrict__ domPtr, int wStride, int bStride) {
  constexpr int R    = 128 / WW;        // conv rows per block
  constexpr int HROW = WW + 2;          // halo row pitch
  constexpr int HW2  = (R + 2) * HROW;  // halo elems per input channel
  constexpr int Ktot = CI * 9;
  constexpr int Wh   = WW >> 1, Hh = HH >> 1;

  __shared__ __align__(16) _Float16 ldsA[NCT * NK * 512]; // [ct][kc][16 m][32 k]
  __shared__ _Float16 halo[CI * HW2];
  __shared__ float    pbuf[16 * 128];

  const int tid = threadIdx.x;
  const int wv  = tid >> 5;
  const int ln  = tid & 31;
  const int b   = blockIdx.x;
  const int r0  = blockIdx.y * R;       // first conv row of this block

  if (domPtr) {
    int d = domPtr[0];
    Wt   += (size_t)d * wStride;
    bias += (size_t)d * bStride;
  }

  // ---- stage weight A-tiles in LDS (K order: kg = ci*9 + (ky*3+kx)) ----
  for (int i = tid; i < NCT * NK * 512; i += 256) {
    int chunk  = i >> 9;
    int within = i & 511;
    int m  = within >> 5;
    int k  = within & 31;
    int ct = chunk / NK;
    int kc = chunk % NK;
    int co = ct * 16 + m;
    int kg = kc * 32 + k;
    float v = 0.0f;
    if (co < Co && kg < Ktot) {
      int ci = kg / 9, tap = kg % 9;
      v = Wt[((size_t)co * CI + ci) * 9 + tap];
    }
    ldsA[i] = (_Float16)v;
  }

  // ---- stage input halo in LDS (each element loaded exactly once) ----
  for (int i = tid; i < CI * HW2; i += 256) {
    int ci  = i / HW2;
    int rem = i % HW2;
    int ly  = rem / HROW;
    int lx  = rem % HROW;
    int gy  = r0 - 1 + ly;
    int gx  = lx - 1;
    _Float16 v = (_Float16)0.0f;
    if (gy >= 0 && gy < HH && gx >= 0 && gx < WW)
      v = (_Float16)in[((size_t)b * CI + ci) * HH * WW + (size_t)gy * WW + gx];
    halo[i] = v;
  }
  __syncthreads();

  v8f acc[NCT];
#pragma unroll
  for (int c = 0; c < NCT; ++c) {
    v8f z = {0.f, 0.f, 0.f, 0.f, 0.f, 0.f, 0.f, 0.f};
    acc[c] = z;
  }

  const int nn = ln & 15;               // WMMA N column / A-row M lane index
  const int gg = ln >> 4;
  const int pix   = wv * 16 + nn;       // this lane's conv pixel
  const int hbase = (pix / WW) * HROW + (pix % WW);  // halo base (tap adds const)

#pragma unroll
  for (int kc = 0; kc < NK; ++kc) {
    // ---- B fragment gathered directly from halo (contiguous k = 16*gg + i) ----
    F16x16 bf;
#pragma unroll
    for (int i = 0; i < 16; ++i) {
      int k  = kc * 32 + 16 * gg + i;
      int kk = (k < Ktot) ? k : 0;      // folds away except in the tail chunk
      int ci = kk / 9, tap = kk % 9;
      _Float16 t = halo[hbase + ci * HW2 + (tap / 3) * HROW + (tap % 3)];
      bf.h[i] = (k < Ktot) ? t : (_Float16)0.0f;
    }
    // ---- A fragments from LDS + WMMA ----
#pragma unroll
    for (int ct = 0; ct < NCT; ++ct) {
      const _Float16* arow = ldsA + (ct * NK + kc) * 512 + nn * 32;
      F16x16 af;
      af.p[0] = *reinterpret_cast<const v8h*>(arow + 16 * gg);
      af.p[1] = *reinterpret_cast<const v8h*>(arow + 16 * gg + 8);
      acc[ct] = __builtin_amdgcn_wmma_f32_16x16x32_f16(
          false, af.v, false, bf.v, (short)0, acc[ct], false, false);
    }
  }

  // ---- epilogue: bias + activation -> LDS -> 2x2 maxpool -> f16 global ----
#pragma unroll
  for (int ct = 0; ct < NCT; ++ct) {
#pragma unroll
    for (int r = 0; r < 8; ++r) {
      int col = r + 8 * gg;             // co within tile
      int co  = ct * 16 + col;
      float v = acc[ct][r] + ((co < Co) ? bias[co] : 0.0f);
      v = (v > 0.0f) ? v : v * slope;
      pbuf[col * 128 + wv * 16 + nn] = v;
    }
    __syncthreads();
    for (int e = tid; e < 512; e += 256) {
      int col = e >> 5;                 // co within tile
      int pp  = e & 31;                 // pooled pixel within block
      int co  = ct * 16 + col;
      if (co < Co) {
        int prow = pp / Wh, pcol = pp % Wh;
        float mx = -3.0e38f;
#pragma unroll
        for (int dy = 0; dy < 2; ++dy)
#pragma unroll
          for (int dx = 0; dx < 2; ++dx)
            mx = fmaxf(mx, pbuf[col * 128 + (2 * prow + dy) * WW + 2 * pcol + dx]);
        out[(((size_t)b * Co + co) * Hh + (r0 >> 1) + prow) * Wh + pcol] =
            (_Float16)mx;
      }
    }
    __syncthreads();
  }
}

// ---------------------------------------------------------------------------
// FC: xcat[b, col0 + n] = act([B,K]f16) @ Wt([12,K]f32)^T + bias, n<12.
// M = 16 batch rows per block; 8 waves split K (stride-8 chunks of 32),
// cross-wave reduce through LDS; writes into fused [B,24] concat buffer.
// Weight row clamped (not masked) for lanes >= 12: their WMMA columns are
// finite garbage and never stored -> no per-lane divergence in the hot loop.
// global_prefetch covers the next chunk of activations/weights.
// ---------------------------------------------------------------------------
__global__ __launch_bounds__(256)
void fc_wmma(const _Float16* __restrict__ act, const float* __restrict__ Wt,
             const float* __restrict__ bias, float* __restrict__ xcat,
             int K, int nK, int col0, int relu,
             const int* __restrict__ domPtr, size_t wStride, int bStride) {
  __shared__ float red[8 * 256];
  const int tid = threadIdx.x, wv = tid >> 5, ln = tid & 31;
  const int b0 = blockIdx.x * 16;
  if (domPtr) {
    int d = domPtr[0];
    Wt   += (size_t)d * wStride;
    bias += (size_t)d * bStride;
  }
  const int mm   = ln & 15;        // A row (batch); also B column (neuron)
  const int gg   = ln >> 4;
  const int wrow = (mm < 12) ? mm : 11;  // clamp: cols 12..15 discarded later

  v8f acc = {0.f, 0.f, 0.f, 0.f, 0.f, 0.f, 0.f, 0.f};
  for (int kc = wv; kc < nK; kc += 8) {
    const _Float16* ar = act + (size_t)(b0 + mm) * K + kc * 32;
    const float*    wr = Wt + (size_t)wrow * K + kc * 32 + 16 * gg;
    __builtin_prefetch(ar + 8 * 32, 0, 1);   // next act chunk (this wave)
    __builtin_prefetch(wr + 8 * 32, 0, 1);   // next weight chunk (this wave)
    F16x16 af;
    af.p[0] = *reinterpret_cast<const v8h*>(ar + 16 * gg);
    af.p[1] = *reinterpret_cast<const v8h*>(ar + 16 * gg + 8);
    F16x16 bf;
#pragma unroll
    for (int i = 0; i < 16; ++i) bf.h[i] = (_Float16)wr[i];
    acc = __builtin_amdgcn_wmma_f32_16x16x32_f16(
        false, af.v, false, bf.v, (short)0, acc, false, false);
  }
#pragma unroll
  for (int r = 0; r < 8; ++r)
    red[wv * 256 + (r + 8 * gg) * 16 + mm] = acc[r];  // [batchM][neuronN]
  __syncthreads();

  float s = 0.0f;
#pragma unroll
  for (int w2 = 0; w2 < 8; ++w2) s += red[w2 * 256 + tid];
  int bm = tid >> 4, n = tid & 15;
  if (n < 12) {
    s += bias[n];
    if (relu) s = fmaxf(s, 0.0f);
    xcat[(size_t)(b0 + bm) * 24 + col0 + n] = s;
  }
}

// ---------------------------------------------------------------------------
// Per-sample task-routed heads (tiny): one thread per sample.
// ---------------------------------------------------------------------------
__global__ __launch_bounds__(256)
void heads_kernel(const float* __restrict__ xcat, const int* __restrict__ tt,
                  const float* __restrict__ H1W, const float* __restrict__ H1b,
                  const float* __restrict__ H2W, const float* __restrict__ H2b,
                  const float* __restrict__ H3W, const float* __restrict__ H3b,
                  float* __restrict__ out, int B) {
  int b = blockIdx.x * blockDim.x + threadIdx.x;
  if (b >= B) return;
  int t = tt[b];
  float x[24];
#pragma unroll
  for (int j = 0; j < 24; ++j) x[j] = xcat[(size_t)b * 24 + j];
  float h1[28];
  const float* w1 = H1W + (size_t)t * 28 * 24;
  for (int i = 0; i < 28; ++i) {
    float s = H1b[t * 28 + i];
#pragma unroll
    for (int j = 0; j < 24; ++j) s += w1[i * 24 + j] * x[j];
    h1[i] = fmaxf(s, 0.0f);
  }
  float h2[14];
  const float* w2 = H2W + (size_t)t * 14 * 28;
  for (int i = 0; i < 14; ++i) {
    float s = H2b[t * 14 + i];
#pragma unroll
    for (int j = 0; j < 28; ++j) s += w2[i * 28 + j] * h1[j];
    h2[i] = fmaxf(s, 0.0f);
  }
  const float* w3 = H3W + (size_t)t * 5 * 14;
  for (int i = 0; i < 5; ++i) {
    float s = H3b[t * 5 + i];
#pragma unroll
    for (int j = 0; j < 14; ++j) s += w3[i * 14 + j] * h2[j];
    out[(size_t)b * 5 + i] = s;
  }
}

// ---------------------------------------------------------------------------
extern "C" void kernel_launch(void* const* d_in, const int* in_sizes, int n_in,
                              void* d_out, int out_size, void* d_ws,
                              size_t ws_size, hipStream_t stream) {
  (void)in_sizes; (void)n_in; (void)out_size; (void)ws_size;
  const float* x_s  = (const float*)d_in[0];
  const float* x_p  = (const float*)d_in[1];
  const int*   tt   = (const int*)d_in[2];
  const int*   dom  = (const int*)d_in[3];
  const float* WinW = (const float*)d_in[4];
  const float* WinB = (const float*)d_in[5];
  const float* WhW  = (const float*)d_in[6];
  const float* WhB  = (const float*)d_in[7];
  const float* WfW  = (const float*)d_in[8];
  const float* WfB  = (const float*)d_in[9];
  const float* PcW  = (const float*)d_in[10];
  const float* PcB  = (const float*)d_in[11];
  const float* PlW  = (const float*)d_in[12];
  const float* PlB  = (const float*)d_in[13];
  const float* H1W  = (const float*)d_in[14];
  const float* H1b  = (const float*)d_in[15];
  const float* H2W  = (const float*)d_in[16];
  const float* H2b  = (const float*)d_in[17];
  const float* H3W  = (const float*)d_in[18];
  const float* H3b  = (const float*)d_in[19];

  const int B = 1024;
  char* ws = (char*)d_ws;
  // Workspace layout (all f16 intermediates; inputs converted in-kernel):
  _Float16* T1 = (_Float16*)(ws + 0);          // [B,36,32,32]  75,497,472 B
  _Float16* T2 = (_Float16*)(ws + 75497472);   // [B,36,16,16]  18,874,368 B
  _Float16* T3 = (_Float16*)(ws + 94371840);   // [B,36, 8, 8]   4,718,592 B
  _Float16* PP = (_Float16*)(ws + 99090432);   // [B,12,32,32]  25,165,824 B
  float*  XCAT = (float*)(ws + 124256256);     // [B,24] f32        98,304 B

  // shared conv1 (f32 in): [B,3,64,64] -> [B,36,32,32]
  conv3x3_relu_pool<3, 1, 3, 64, 64, float><<<dim3(B, 32), 256, 0, stream>>>(
      x_s, WinW, WinB, T1, 36, 0.0f, nullptr, 0, 0);
  // shared hid1: [B,36,32,32] -> [B,36,16,16]
  conv3x3_relu_pool<3, 11, 36, 32, 32, _Float16><<<dim3(B, 8), 256, 0, stream>>>(
      T1, WhW, WhB, T2, 36, 0.0f, nullptr, 0, 0);
  // shared hid2: [B,36,16,16] -> [B,36,8,8]
  conv3x3_relu_pool<3, 11, 36, 16, 16, _Float16><<<dim3(B, 2), 256, 0, stream>>>(
      T2, WhW, WhB, T3, 36, 0.0f, nullptr, 0, 0);
  // private conv (f32 in, leaky 0.001): [B,3,64,64] -> [B,12,32,32]
  conv3x3_relu_pool<1, 1, 3, 64, 64, float><<<dim3(B, 32), 256, 0, stream>>>(
      x_p, PcW, PcB, PP, 12, 0.001f, dom, 12 * 3 * 9, 12);

  // shared FC (+ReLU) -> xcat[:, 0:12]
  fc_wmma<<<64, 256, 0, stream>>>(T3, WfW, WfB, XCAT, 2304, 72, 0, 1,
                                  nullptr, (size_t)0, 0);
  // private FC -> xcat[:, 12:24]
  fc_wmma<<<64, 256, 0, stream>>>(PP, PlW, PlB, XCAT, 12288, 384, 12, 0,
                                  dom, (size_t)12 * 12288, 12);

  // per-sample routed heads -> out [1024,5]
  heads_kernel<<<4, 256, 0, stream>>>(XCAT, tt, H1W, H1b, H2W, H2b, H3W, H3b,
                                      (float*)d_out, B);
}
